// NNSensorResponse_687194768378
// MI455X (gfx1250) — matrix operand order repaired and checked by reading, compiled, not run on hardware
//
#include <hip/hip_runtime.h>
#include <hip/hip_bf16.h>

// ---------------------------------------------------------------------------
// NNSensorResponse for MI455X (gfx1250, wave32, WMMA, async-to-LDS).
//
//   k_hidden   : per-electron 3->128 MLP hidden layers (f32 VALU) -> bf16
//   k_wt       : W2 [128,1024] f32 -> W2t [1024,128] bf16 (K-contiguous rows)
//   k_exp      : Gaussian table Et [T=1024][K=32000] bf16
//   k_response : WMMA bf16 GEMM (K=128) + sigmoid*amp*mask epilogue
//                -> Rt [S=1024][K=32000] bf16 (pre-transposed = A operand)
//   k_waveform : C[s,t] = sum_k Rt[s,k]*Et[t,k], 67 GFLOP, split-K=4 into
//                f32 partials (deterministic), then k_reduce sums them.
//                Staging uses GLOBAL_LOAD_ASYNC_TO_LDS_B128 + ASYNCcnt
//                double buffering (CDNA5 async path, ISA ch.10.7).
//
// Wave32: 128-thread blocks = 4 waves; each wave owns a 32x32 quadrant of a
// 64x64 C tile. Rt + Et = 131 MB bf16 -> resident in 192 MB L2, so the big
// GEMM is v_wmma_f32_16x16x32_bf16 throughput bound, not HBM bound.
// ---------------------------------------------------------------------------

typedef __bf16 bf16_t;
typedef __attribute__((ext_vector_type(16))) __bf16        v16bf;
typedef __attribute__((ext_vector_type(8)))  float         v8f;
typedef __attribute__((ext_vector_type(8)))  unsigned int  v8u;

#define KE   32000   // B*N electrons (4*8000)
#define HID  128
#define SN   1024    // sensors
#define TT   1024    // waveform ticks
#define KSPLIT 4     // split-K factor for the dominant GEMM
#define INV_SQRT_2PI 0.3989422804f

__device__ __forceinline__ unsigned short f2bf(float f) {
  bf16_t h = (bf16_t)f;                       // RNE convert
  return __builtin_bit_cast(unsigned short, h);
}
__device__ __forceinline__ unsigned int pack2(float lo, float hi) {
  return (unsigned int)f2bf(lo) | ((unsigned int)f2bf(hi) << 16);
}
__device__ __forceinline__ v16bf make16(uint4 a, uint4 b) {
  v8u u;
  u[0] = a.x; u[1] = a.y; u[2] = a.z; u[3] = a.w;
  u[4] = b.x; u[5] = b.y; u[6] = b.z; u[7] = b.w;
  return __builtin_bit_cast(v16bf, u);
}
// A-matrix 16x32 bf16 fragment (ISA 7.12.2): lane = m + 16*hi.
// v0..3 hold K = 8*hi+0..7 ; v4..7 hold K = 16+8*hi+0..7 (2 bf16 / vgpr).
// rowp points at 32 contiguous bf16 of row m (K-major).
__device__ __forceinline__ v16bf fragA(const unsigned short* rowp, int hi) {
  uint4 a = *(const uint4*)(rowp + 8 * hi);
  uint4 b = *(const uint4*)(rowp + 16 + 8 * hi);
  return make16(a, b);
}
// B-matrix 32x16 bf16 fragment: lane = n + 16*hi; v0..3 hold K = 8*hi+0..7 ; v4..7 hold K = 16+8*hi+0..7.
__device__ __forceinline__ v16bf fragB(const unsigned short* rowp, int hi) {
  uint4 a = *(const uint4*)(rowp + 8 * hi);
  uint4 b = *(const uint4*)(rowp + 16 + 8 * hi);
  return make16(a, b);
}
#define WMMA_BF16(A, Bm, C) \
  __builtin_amdgcn_wmma_f32_16x16x32_bf16(false, (A), false, (Bm), (short)0, (C), false, false)

// CDNA5 async copy: LDS[lds_off] = MEM[gaddr], 16 bytes, tracked by ASYNCcnt.
// lds_off = low 32 bits of the generic pointer (flat-LDS aperture rule).
__device__ __forceinline__ void async_ld_b128(unsigned lds_off, const void* gptr) {
  asm volatile("global_load_async_to_lds_b128 %0, %1, off"
               :: "v"(lds_off), "v"((unsigned long long)(size_t)gptr)
               : "memory");
}
#define WAIT_ASYNC(n) asm volatile("s_wait_asynccnt " #n ::: "memory")
__device__ __forceinline__ unsigned lds_addr(const void* p) {
  return (unsigned)(size_t)p;                 // addr[31:0] = LDS byte offset
}

// ---------------------------------------------------------------------------
// Kernel 1: hidden layers. One thread per electron; 3 FMAs per hidden unit.
// ---------------------------------------------------------------------------
__global__ __launch_bounds__(256) void k_hidden(
    const float* __restrict__ X, const float* __restrict__ Wp1, const float* __restrict__ bp1,
    const float* __restrict__ Wa1, const float* __restrict__ ba1,
    unsigned int* __restrict__ Hp, unsigned int* __restrict__ Ha) {
  int e = blockIdx.x * blockDim.x + threadIdx.x;
  if (e >= KE) return;
  float x0 = X[e * 3 + 0], x1 = X[e * 3 + 1], x2 = X[e * 3 + 2];
#pragma unroll 4
  for (int h = 0; h < HID; h += 2) {
    float p0 = fmaf(x0, Wp1[h],     fmaf(x1, Wp1[HID + h],     fmaf(x2, Wp1[2 * HID + h],     bp1[h])));
    float p1 = fmaf(x0, Wp1[h + 1], fmaf(x1, Wp1[HID + h + 1], fmaf(x2, Wp1[2 * HID + h + 1], bp1[h + 1])));
    float a0 = fmaf(x0, Wa1[h],     fmaf(x1, Wa1[HID + h],     fmaf(x2, Wa1[2 * HID + h],     ba1[h])));
    float a1 = fmaf(x0, Wa1[h + 1], fmaf(x1, Wa1[HID + h + 1], fmaf(x2, Wa1[2 * HID + h + 1], ba1[h + 1])));
    p0 = fmaxf(p0, 0.0f); p1 = fmaxf(p1, 0.0f);
    a0 = fmaxf(a0, 0.0f); a1 = fmaxf(a1, 0.0f);
    Hp[(e * HID + h) >> 1] = pack2(p0, p1);
    Ha[(e * HID + h) >> 1] = pack2(a0, a1);
  }
}

// ---------------------------------------------------------------------------
// Kernel 2: W2 [HID,SN] f32 -> W2t [SN,HID] bf16 (rows K-contiguous).
// ---------------------------------------------------------------------------
__global__ __launch_bounds__(256) void k_wt(
    const float* __restrict__ Wp2, const float* __restrict__ Wa2,
    unsigned short* __restrict__ Wp2t, unsigned short* __restrict__ Wa2t) {
  int idx = blockIdx.x * blockDim.x + threadIdx.x;
  if (idx >= SN * HID) return;
  int s = idx / HID, h = idx % HID;
  Wp2t[idx] = f2bf(Wp2[h * SN + s]);
  Wa2t[idx] = f2bf(Wa2[h * SN + s]);
}

// ---------------------------------------------------------------------------
// Kernel 3: Gaussian table Et[t][k] = exp(-(t-z_k)^2/(2s^2)) * c, bf16 packed.
// ---------------------------------------------------------------------------
__global__ __launch_bounds__(256) void k_exp(
    const float* __restrict__ z, const float* __restrict__ sig,
    unsigned int* __restrict__ Et) {
  int idx = blockIdx.x * blockDim.x + threadIdx.x;   // over TT * (KE/2)
  int t  = idx / (KE / 2);
  int kp = idx - t * (KE / 2);
  int k  = kp * 2;
  float sigma = fabsf(sig[0]);
  float inv2  = 1.0f / (2.0f * sigma * sigma);
  float coef  = INV_SQRT_2PI / sigma;
  float d0 = (float)t - z[k];
  float d1 = (float)t - z[k + 1];
  float e0 = __expf(-d0 * d0 * inv2) * coef;
  float e1 = __expf(-d1 * d1 * inv2) * coef;
  Et[idx] = pack2(e0, e1);
}

// ---------------------------------------------------------------------------
// Kernel 4: Rt[s,ke] = sigmoid(Hp@Wp2 + bp2) * (Ha@Wa2 + ba2) * mask, bf16.
// GEMM M=SN, N=KE, K=HID=128. 64x64 C tile / 128-thread block (4 waves,
// each 32x32). Full K extent staged in LDS once (4 x 16 KB), one barrier.
// ---------------------------------------------------------------------------
__global__ __launch_bounds__(128) void k_response(
    const unsigned short* __restrict__ Hp, const unsigned short* __restrict__ Ha,
    const unsigned short* __restrict__ Wp2t, const unsigned short* __restrict__ Wa2t,
    const float* __restrict__ bp2, const float* __restrict__ ba2,
    const float* __restrict__ mask, unsigned short* __restrict__ Rt) {
  __shared__ __align__(16) unsigned short sAp[64 * HID], sAa[64 * HID];
  __shared__ __align__(16) unsigned short sBp[64 * HID], sBa[64 * HID];
  const int e0 = blockIdx.x * 64;
  const int s0 = blockIdx.y * 64;
  const int tid  = threadIdx.x;               // 128 threads = 4 waves (wave32)
  const int wave = tid >> 5, lane = tid & 31;
  const int m_off = (wave >> 1) * 32;         // {0,32}
  const int n_off = (wave & 1) * 32;          // {0,32}
  const int ml = lane & 15, hi = lane >> 4;

  // Stage full 64x128 strips: 8192 ushort = 1024 uint4 per array,
  // 128 threads -> 8 uint4 each.
#pragma unroll
  for (int j = 0; j < 8; ++j) {
    const int idx = tid + 128 * j;            // 0..1023
    const int r = idx >> 3, c = (idx & 7) * 16;
    *(uint4*)&sAp[r * HID + c] = *(const uint4*)&Wp2t[(s0 + r) * HID + c];
    *(uint4*)&sAa[r * HID + c] = *(const uint4*)&Wa2t[(s0 + r) * HID + c];
    *(uint4*)&sBp[r * HID + c] = *(const uint4*)&Hp[(size_t)(e0 + r) * HID + c];
    *(uint4*)&sBa[r * HID + c] = *(const uint4*)&Ha[(size_t)(e0 + r) * HID + c];
  }
  __syncthreads();

  v8f accP[2][2] = {};
  v8f accA[2][2] = {};
#pragma unroll
  for (int kb = 0; kb < HID; kb += 32) {
    v16bf ap0 = fragA(&sAp[(m_off + ml) * HID + kb], hi);
    v16bf ap1 = fragA(&sAp[(m_off + 16 + ml) * HID + kb], hi);
    v16bf aa0 = fragA(&sAa[(m_off + ml) * HID + kb], hi);
    v16bf aa1 = fragA(&sAa[(m_off + 16 + ml) * HID + kb], hi);
    v16bf bp0 = fragB(&sBp[(n_off + ml) * HID + kb], hi);
    v16bf bp1f = fragB(&sBp[(n_off + 16 + ml) * HID + kb], hi);
    v16bf ba0 = fragB(&sBa[(n_off + ml) * HID + kb], hi);
    v16bf ba1f = fragB(&sBa[(n_off + 16 + ml) * HID + kb], hi);
    accP[0][0] = WMMA_BF16(ap0, bp0,  accP[0][0]);
    accP[0][1] = WMMA_BF16(ap0, bp1f, accP[0][1]);
    accP[1][0] = WMMA_BF16(ap1, bp0,  accP[1][0]);
    accP[1][1] = WMMA_BF16(ap1, bp1f, accP[1][1]);
    accA[0][0] = WMMA_BF16(aa0, ba0,  accA[0][0]);
    accA[0][1] = WMMA_BF16(aa0, ba1f, accA[0][1]);
    accA[1][0] = WMMA_BF16(aa1, ba0,  accA[1][0]);
    accA[1][1] = WMMA_BF16(aa1, ba1f, accA[1][1]);
  }

#pragma unroll
  for (int mt = 0; mt < 2; ++mt)
#pragma unroll
    for (int nt = 0; nt < 2; ++nt) {
      const int sBase = s0 + m_off + 16 * mt;
      const int ke    = e0 + n_off + 16 * nt + ml;   // n = lane&15
      const float mk  = mask[ke];
#pragma unroll
      for (int r = 0; r < 8; ++r) {                  // m = r + 8*hi
        const int s = sBase + r + 8 * hi;
        float p   = accP[mt][nt][r] + bp2[s];
        float amp = accA[mt][nt][r] + ba2[s];
        float sg  = 1.0f / (1.0f + __expf(-p));
        Rt[(size_t)s * KE + ke] = f2bf(sg * amp * mk);
      }
    }
}

// ---------------------------------------------------------------------------
// Kernel 5: dominant GEMM, split-K. Pbuf[z][s][t] = sum_{k in slice z}.
// Grid (TT/64, SN/64, KSPLIT); 128-thread block = 4 waves, each 32x32 of a
// 64x64 C tile. K-step 32 staged with GLOBAL_LOAD_ASYNC_TO_LDS_B128 into a
// 2-deep LDS double buffer; ASYNCcnt retires in order, so s_wait_asynccnt 4
// after issuing the next tile's 4 loads guarantees the current tile is ready.
// ---------------------------------------------------------------------------
__global__ __launch_bounds__(128) void k_waveform(
    const unsigned short* __restrict__ Rt, const unsigned short* __restrict__ Et,
    float* __restrict__ Pbuf) {
  __shared__ __align__(16) unsigned short sA[2][64 * 32], sB[2][64 * 32];
  const int t0 = blockIdx.x * 64;
  const int s0 = blockIdx.y * 64;
  const int ks = blockIdx.z * (KE / KSPLIT);
  const int tid  = threadIdx.x;               // 128 threads = 4 waves (wave32)
  const int wave = tid >> 5, lane = tid & 31;
  const int m_off = (wave >> 1) * 32;         // {0,32}
  const int n_off = (wave & 1) * 32;          // {0,32}
  const int ml = lane & 15, hi = lane >> 4;
  // Staging map: 64 rows x 32 cols = 256 uint4 per tile; 2 uint4 per thread.
  const int r0 = tid >> 1,         c0 = (tid & 1) * 16;
  const int r1 = (tid + 128) >> 1, c1 = ((tid + 128) & 1) * 16;
  const unsigned short* Ag0 = Rt + (size_t)(s0 + r0) * KE + ks + c0;
  const unsigned short* Ag1 = Rt + (size_t)(s0 + r1) * KE + ks + c1;
  const unsigned short* Bg0 = Et + (size_t)(t0 + r0) * KE + ks + c0;
  const unsigned short* Bg1 = Et + (size_t)(t0 + r1) * KE + ks + c1;
  const unsigned la0 = lds_addr(&sA[0][r0 * 32 + c0]);
  const unsigned la1 = lds_addr(&sA[0][r1 * 32 + c1]);
  const unsigned lb0 = lds_addr(&sB[0][r0 * 32 + c0]);
  const unsigned lb1 = lds_addr(&sB[0][r1 * 32 + c1]);
  const unsigned bufB = 64 * 32 * 2;          // bytes per buffer plane

  v8f acc[2][2] = {};

  // Prologue: stage K-step 0 into buffer 0.
  async_ld_b128(la0, Ag0); async_ld_b128(la1, Ag1);
  async_ld_b128(lb0, Bg0); async_ld_b128(lb1, Bg1);

  const int nsteps = (KE / KSPLIT) / 32;      // 250
  for (int i = 0; i < nsteps; ++i) {
    const int cur = i & 1;
    if (i + 1 < nsteps) {
      const unsigned nb = (cur ^ 1) * bufB;
      const int kb2 = (i + 1) * 32;
      async_ld_b128(la0 + nb, Ag0 + kb2); async_ld_b128(la1 + nb, Ag1 + kb2);
      async_ld_b128(lb0 + nb, Bg0 + kb2); async_ld_b128(lb1 + nb, Bg1 + kb2);
      WAIT_ASYNC(4);                          // current buffer's 4 loads done
    } else {
      WAIT_ASYNC(0);
    }
    __syncthreads();                          // all waves' loads visible
    v16bf a0 = fragA(&sA[cur][(m_off + ml) * 32], hi);
    v16bf a1 = fragA(&sA[cur][(m_off + 16 + ml) * 32], hi);
    v16bf b0 = fragB(&sB[cur][(n_off + ml) * 32], hi);
    v16bf b1 = fragB(&sB[cur][(n_off + 16 + ml) * 32], hi);
    acc[0][0] = WMMA_BF16(a0, b0, acc[0][0]);
    acc[0][1] = WMMA_BF16(a0, b1, acc[0][1]);
    acc[1][0] = WMMA_BF16(a1, b0, acc[1][0]);
    acc[1][1] = WMMA_BF16(a1, b1, acc[1][1]);
    __syncthreads();                          // reads done before buf reuse
  }

  float* P = Pbuf + (size_t)blockIdx.z * SN * TT;
#pragma unroll
  for (int mt = 0; mt < 2; ++mt)
#pragma unroll
    for (int nt = 0; nt < 2; ++nt) {
      const int sBase = s0 + m_off + 16 * mt;
      const int tIdx  = t0 + n_off + 16 * nt + ml;
#pragma unroll
      for (int r = 0; r < 8; ++r)
        P[(size_t)(sBase + r + 8 * hi) * TT + tIdx] = acc[mt][nt][r];
    }
}

// ---------------------------------------------------------------------------
// Kernel 6: sum the KSPLIT partial planes into the final [S,T] f32 output.
// ---------------------------------------------------------------------------
__global__ __launch_bounds__(256) void k_reduce(
    const float* __restrict__ Pbuf, float* __restrict__ out) {
  int i = blockIdx.x * blockDim.x + threadIdx.x;
  if (i >= SN * TT) return;
  const int M = SN * TT;
  out[i] = (Pbuf[i] + Pbuf[i + M]) + (Pbuf[i + 2 * M] + Pbuf[i + 3 * M]);
}

// ---------------------------------------------------------------------------
extern "C" void kernel_launch(void* const* d_in, const int* in_sizes, int n_in,
                              void* d_out, int out_size, void* d_ws, size_t ws_size,
                              hipStream_t stream) {
  const float* X    = (const float*)d_in[0];   // [B,N,3]
  const float* z    = (const float*)d_in[1];   // [B,N]
  const float* mask = (const float*)d_in[2];   // [B,N,1]
  const float* Wp1  = (const float*)d_in[3];   // [3,128]
  const float* bp1  = (const float*)d_in[4];   // [128]
  const float* Wp2  = (const float*)d_in[5];   // [128,1024]
  const float* bp2  = (const float*)d_in[6];   // [1024]
  const float* Wa1  = (const float*)d_in[7];
  const float* ba1  = (const float*)d_in[8];
  const float* Wa2  = (const float*)d_in[9];
  const float* ba2  = (const float*)d_in[10];
  const float* sig  = (const float*)d_in[11];  // [1]
  float* out = (float*)d_out;                  // [1024,1024] f32

  // Workspace carve-up (all 16B-aligned; ~148 MB bf16 + 16 MB f32 partials):
  unsigned short* Rt   = (unsigned short*)d_ws;           // [SN, KE]  bf16
  unsigned short* Et   = Rt   + (size_t)SN * KE;          // [TT, KE]  bf16
  unsigned short* Hp   = Et   + (size_t)TT * KE;          // [KE, HID] bf16
  unsigned short* Ha   = Hp   + (size_t)KE * HID;         // [KE, HID] bf16
  unsigned short* Wp2t = Ha   + (size_t)KE * HID;         // [SN, HID] bf16
  unsigned short* Wa2t = Wp2t + (size_t)SN * HID;         // [SN, HID] bf16
  float*          Pbuf = (float*)(Wa2t + (size_t)SN * HID); // [KSPLIT, SN, TT]

  k_hidden<<<KE / 256, 256, 0, stream>>>(X, Wp1, bp1, Wa1, ba1,
                                         (unsigned int*)Hp, (unsigned int*)Ha);
  k_wt<<<(SN * HID) / 256, 256, 0, stream>>>(Wp2, Wa2, Wp2t, Wa2t);
  k_exp<<<(TT * (KE / 2)) / 256, 256, 0, stream>>>(z, sig, (unsigned int*)Et);
  k_response<<<dim3(KE / 64, SN / 64), 128, 0, stream>>>(Hp, Ha, Wp2t, Wa2t,
                                                         bp2, ba2, mask, Rt);
  k_waveform<<<dim3(TT / 64, SN / 64, KSPLIT), 128, 0, stream>>>(Rt, Et, Pbuf);
  k_reduce<<<(SN * TT) / 256, 256, 0, stream>>>(Pbuf, out);

  (void)in_sizes; (void)n_in; (void)out_size; (void)ws_size;
}
